// OuterProductMean_48155173322943
// MI455X (gfx1250) — matrix operand order repaired and checked by reading, not compiled, not run on hardware
//
#include <hip/hip_runtime.h>

typedef __attribute__((ext_vector_type(16))) __bf16 v16bf;
typedef __attribute__((ext_vector_type(8)))  __bf16 v8bf;
typedef __attribute__((ext_vector_type(8)))  float  v8f;

#define DD   128   // D
#define HH   64    // H
#define EE   64    // E
#define ROWS 256   // rows per block tile
#define LSTR 136   // LDS row stride in bf16 elements (pad: 272B -> conflict-free)
#define LN_EPS 1e-5f

// ---------------------------------------------------------------------------
// Kernel 1: per-tile partial sums of layernorm(x) rows -> partials[b][tile][D]
// Deterministic (no atomics): fixed reduction order in kernel 2.
// ---------------------------------------------------------------------------
__global__ void k1_xnsum_partials(const float* __restrict__ x,
                                  const float* __restrict__ ln_w,
                                  const float* __restrict__ ln_b,
                                  float* __restrict__ partials,
                                  int N, int tilesPerBatch) {
  const int tile = blockIdx.x % tilesPerBatch;
  const int b    = blockIdx.x / tilesPerBatch;
  const int lane = threadIdx.x & 31;
  const int wave = threadIdx.x >> 5;

  const float* xb = x + ((size_t)b * N + (size_t)tile * ROWS) * DD;
  const float4 wv = ((const float4*)ln_w)[lane];
  const float4 bv = ((const float4*)ln_b)[lane];

  float4 acc = make_float4(0.f, 0.f, 0.f, 0.f);
  const int r0 = wave * 32;
#pragma unroll 4
  for (int r = r0; r < r0 + 32; ++r) {
    float4 v = ((const float4*)(xb + (size_t)r * DD))[lane];
    float s = v.x + v.y + v.z + v.w;
#pragma unroll
    for (int m = 16; m >= 1; m >>= 1) s += __shfl_xor(s, m, 32);
    const float mu = s * (1.0f / 128.0f);
    const float dx = v.x - mu, dy = v.y - mu, dz = v.z - mu, dw = v.w - mu;
    float q = dx * dx + dy * dy + dz * dz + dw * dw;
#pragma unroll
    for (int m = 16; m >= 1; m >>= 1) q += __shfl_xor(q, m, 32);
    const float rs = rsqrtf(q * (1.0f / 128.0f) + LN_EPS);
    acc.x += dx * rs * wv.x + bv.x;
    acc.y += dy * rs * wv.y + bv.y;
    acc.z += dz * rs * wv.z + bv.z;
    acc.w += dw * rs * wv.w + bv.w;
  }

  __shared__ float red[8][DD];
  red[wave][lane * 4 + 0] = acc.x;
  red[wave][lane * 4 + 1] = acc.y;
  red[wave][lane * 4 + 2] = acc.z;
  red[wave][lane * 4 + 3] = acc.w;
  __syncthreads();
  if (threadIdx.x < DD) {
    float s = 0.f;
#pragma unroll
    for (int w = 0; w < 8; ++w) s += red[w][threadIdx.x];
    partials[((size_t)b * tilesPerBatch + tile) * DD + threadIdx.x] = s;
  }
}

// ---------------------------------------------------------------------------
// Kernel 2: per batch b:
//   xnsum  = sum of partials (fixed order)
//   lm[h]  = (xnsum/N) . Wl[:,h] + bl[h]
//   WfT[b][e][d] = sum_h Wr[d,h] * lm[h] * Wo[h,e]          (bf16, transposed)
//   bfB[b][e]    = sum_h br[h]  * lm[h] * Wo[h,e] + bo[e]
// One 128-thread block per batch (tiny: ~1 MFLOP/batch).
// ---------------------------------------------------------------------------
__global__ void k2_fuse(const float* __restrict__ partials,
                        const float* __restrict__ Wl, const float* __restrict__ bl,
                        const float* __restrict__ Wr, const float* __restrict__ br,
                        const float* __restrict__ Wo, const float* __restrict__ bo,
                        __bf16* __restrict__ WfT, float* __restrict__ bfB,
                        int N, int tilesPerBatch) {
  const int b   = blockIdx.x;
  const int tid = threadIdx.x;  // 128 threads
  __shared__ float xnsum[DD];
  __shared__ float lm[HH];

  float s = 0.f;
  const float* p = partials + (size_t)b * tilesPerBatch * DD + tid;
  for (int t = 0; t < tilesPerBatch; ++t) s += p[(size_t)t * DD];
  xnsum[tid] = s;
  __syncthreads();

  if (tid < HH) {
    float a = 0.f;
#pragma unroll 8
    for (int d = 0; d < DD; ++d) a += xnsum[d] * Wl[d * HH + tid];
    lm[tid] = a / (float)N + bl[tid];
  }
  __syncthreads();

  __bf16* wfb = WfT + (size_t)b * EE * DD;
  for (int idx = tid; idx < EE * DD; idx += 128) {
    const int e = idx >> 7;      // / DD
    const int d = idx & (DD - 1);
    float a = 0.f;
#pragma unroll 8
    for (int h = 0; h < HH; ++h) a += Wr[d * HH + h] * lm[h] * Wo[h * EE + e];
    wfb[idx] = (__bf16)a;        // WfT[e][d]
  }
  if (tid < EE) {
    float a = 0.f;
#pragma unroll 8
    for (int h = 0; h < HH; ++h) a += br[h] * lm[h] * Wo[h * EE + tid];
    bfB[b * EE + tid] = a + bo[tid];
  }
}

// ---------------------------------------------------------------------------
// Kernel 3: out[b,n,:] = layernorm(x[b,n,:]) @ Wf[b] + bfB[b]
// 256 threads = 8 waves; 256-row tile; bf16 WMMA 16x16x32, f32 accumulate.
// ---------------------------------------------------------------------------
__global__ void __launch_bounds__(256)
k3_gemm(const float* __restrict__ x,
        const float* __restrict__ ln_w, const float* __restrict__ ln_b,
        const __bf16* __restrict__ WfT, const float* __restrict__ bfB,
        float* __restrict__ out, int N, int tilesPerBatch) {
  const int tile = blockIdx.x % tilesPerBatch;
  const int b    = blockIdx.x / tilesPerBatch;
  const int lane = threadIdx.x & 31;
  const int wave = threadIdx.x >> 5;
  const int lmod = lane & 15;
  const int half = lane >> 4;   // 0: lanes 0-15, 1: lanes 16-31

  __shared__ __align__(16) __bf16 sA[ROWS * LSTR];

  // ---- Phase 1: layernorm 32 rows per wave into LDS (bf16) ----
  const float* xb = x + ((size_t)b * N + (size_t)tile * ROWS) * DD;
  const float4 wv = ((const float4*)ln_w)[lane];
  const float4 bv = ((const float4*)ln_b)[lane];
  const int r0 = wave * 32;
#pragma unroll 4
  for (int r = r0; r < r0 + 32; ++r) {
    float4 v = ((const float4*)(xb + (size_t)r * DD))[lane];
    float s = v.x + v.y + v.z + v.w;
#pragma unroll
    for (int m = 16; m >= 1; m >>= 1) s += __shfl_xor(s, m, 32);
    const float mu = s * (1.0f / 128.0f);
    const float dx = v.x - mu, dy = v.y - mu, dz = v.z - mu, dw = v.w - mu;
    float q = dx * dx + dy * dy + dz * dz + dw * dw;
#pragma unroll
    for (int m = 16; m >= 1; m >>= 1) q += __shfl_xor(q, m, 32);
    const float rs = rsqrtf(q * (1.0f / 128.0f) + LN_EPS);
    __bf16* dst = &sA[r * LSTR + lane * 4];
    dst[0] = (__bf16)(dx * rs * wv.x + bv.x);
    dst[1] = (__bf16)(dy * rs * wv.y + bv.y);
    dst[2] = (__bf16)(dz * rs * wv.z + bv.z);
    dst[3] = (__bf16)(dw * rs * wv.w + bv.w);
  }
  __syncthreads();

  // ---- Phase 2: WMMA. Wave handles rows [wave*32, wave*32+32), all 64 cols.
  v8f acc[2][4] = {};
  const __bf16* wf = WfT + (size_t)b * EE * DD;
  const int mrow = wave * 32;

#pragma unroll
  for (int kt = 0; kt < 4; ++kt) {
    // A fragment: 16-bit A layout: lane half 0 -> K chunks [0..7],[16..23];
    // half 1 -> [8..15],[24..31] within this kt's 32-wide K slice.
    const int ka = kt * 32 + half * 8;
    v16bf afrag[2];
#pragma unroll
    for (int mt = 0; mt < 2; ++mt) {
      const __bf16* ap = &sA[(mrow + mt * 16 + lmod) * LSTR + ka];
      v8bf lo = *(const v8bf*)ap;
      v8bf hi = *(const v8bf*)(ap + 16);
      afrag[mt] = __builtin_shufflevector(lo, hi, 0, 1, 2, 3, 4, 5, 6, 7,
                                                  8, 9, 10, 11, 12, 13, 14, 15);
    }
    // B fragment: column n = nt*16+lmod, K chunks [half*8..+7],[16+half*8..+7]
    // within kt*32 slice (same as A); WfT is [E][D] so K is contiguous.
    const int kb = kt * 32 + half * 8;
#pragma unroll
    for (int nt = 0; nt < 4; ++nt) {
      const __bf16* bp = wf + (size_t)(nt * 16 + lmod) * DD + kb;
      v8bf lo = *(const v8bf*)bp;
      v8bf hi = *(const v8bf*)(bp + 16);
      v16bf bfrag = __builtin_shufflevector(lo, hi, 0, 1, 2, 3, 4, 5, 6, 7,
                                                    8, 9, 10, 11, 12, 13, 14, 15);
      acc[0][nt] = __builtin_amdgcn_wmma_f32_16x16x32_bf16(
          false, afrag[0], false, bfrag, (short)0, acc[0][nt], false, false);
      acc[1][nt] = __builtin_amdgcn_wmma_f32_16x16x32_bf16(
          false, afrag[1], false, bfrag, (short)0, acc[1][nt], false, false);
    }
  }

  // ---- Phase 3: bias + store. C/D layout: VGPR j, lanes 0-15 -> M=j,
  // lanes 16-31 -> M=8+j; N = lane%16.
  const size_t rowBase = (size_t)b * N + (size_t)tile * ROWS + mrow;
#pragma unroll
  for (int mt = 0; mt < 2; ++mt) {
#pragma unroll
    for (int nt = 0; nt < 4; ++nt) {
      const int col = nt * 16 + lmod;
      const float bias = bfB[b * EE + col];
#pragma unroll
      for (int j = 0; j < 8; ++j) {
        const int m = mt * 16 + half * 8 + j;
        out[(rowBase + m) * EE + col] = acc[mt][nt][j] + bias;
      }
    }
  }
}

// ---------------------------------------------------------------------------
extern "C" void kernel_launch(void* const* d_in, const int* in_sizes, int n_in,
                              void* d_out, int out_size, void* d_ws, size_t ws_size,
                              hipStream_t stream) {
  const float* x    = (const float*)d_in[0];
  const float* ln_w = (const float*)d_in[1];
  const float* ln_b = (const float*)d_in[2];
  const float* Wl   = (const float*)d_in[3];
  const float* bl   = (const float*)d_in[4];
  const float* Wr   = (const float*)d_in[5];
  const float* br   = (const float*)d_in[6];
  const float* Wo   = (const float*)d_in[7];
  const float* bo   = (const float*)d_in[8];
  float* out = (float*)d_out;

  const int B = 16;                          // reference batch
  const int BN = in_sizes[0] / DD;           // B*N
  const int N = BN / B;                      // 8192
  const int tiles = N / ROWS;                // 32

  char* ws = (char*)d_ws;
  float*  partials = (float*)ws;                                   // B*tiles*D f32
  size_t  off1 = (size_t)B * tiles * DD * sizeof(float);
  __bf16* WfT = (__bf16*)(ws + off1);                              // B*E*D bf16
  size_t  off2 = off1 + (size_t)B * EE * DD * 2;
  float*  bfB = (float*)(ws + off2);                               // B*E f32

  k1_xnsum_partials<<<B * tiles, 256, 0, stream>>>(x, ln_w, ln_b, partials, N, tiles);
  k2_fuse<<<B, 128, 0, stream>>>(partials, Wl, bl, Wr, br, Wo, bo, WfT, bfB, N, tiles);
  k3_gemm<<<B * tiles, 256, 0, stream>>>(x, ln_w, ln_b, WfT, bfB, out, N, tiles);
}
